// MLPAttention_54795192762707
// MI455X (gfx1250) — matrix operand order repaired and checked by reading, not compiled, not run on hardware
//
#include <hip/hip_runtime.h>

// ---------------------------------------------------------------------------
// MLP-attention for MI455X (gfx1250), wave32 + WMMA bf16.
//
//   scores[q,k] = W2 . relu( W1d@dist[:,q,k] + KBt[k] + QB[q] ) + b2
//   (QB = W1q@query + b1, KBt = (W1k@keys)^T  -- precomputed, L2-resident)
//   then global-min floor, mask add, softmax over k, out = prob @ value^T.
//
// Hot loop: V_WMMA_F32_16X16X32_BF16, f32 accumulate. dist (134MB) is the
// only large HBM stream -> kernel is HBM-bound (~7us floor @ 23.3 TB/s).
// Round 4: f32->bf16 pair conversion via v_cvt_pk_bf16_f32 inline asm
// (clang builtin absent; manual RNE chains were ~5 VALU/pair). Async
// global->LDS A-frag staging + global_prefetch retained from round 3.
// ---------------------------------------------------------------------------

#define C_DIM 128
#define NQ_ 8
#define NK_ 32768
#define LL_ ((size_t)NQ_ * NK_)   // 262144

// Flip to 0 if the assembler rejects the async mnemonic.
#define USE_ASYNC_LDS 1
// Flip to 0 if the assembler rejects v_cvt_pk_bf16_f32.
#define USE_ASM_CVT 1

typedef __bf16          v16bf __attribute__((ext_vector_type(16)));
typedef unsigned short  v16us __attribute__((ext_vector_type(16)));
typedef unsigned int    v8ui  __attribute__((ext_vector_type(8)));
typedef float           v8f   __attribute__((ext_vector_type(8)));

// f32 -> bf16, round-to-nearest-even (pure integer ops, always compiles)
__device__ __forceinline__ unsigned short f2bf(float f) {
    unsigned u = __float_as_uint(f);
    unsigned r = u + 0x7FFFu + ((u >> 16) & 1u);
    return (unsigned short)(r >> 16);
}

// Pack two f32 into one dword of 2 x bf16 (lo = a, hi = b).
__device__ __forceinline__ unsigned pack_bf16x2(float a, float b) {
#if USE_ASM_CVT
    unsigned r;
    asm("v_cvt_pk_bf16_f32 %0, %1, %2" : "=v"(r) : "v"(a), "v"(b));
    return r;
#else
    return (unsigned)f2bf(a) | ((unsigned)f2bf(b) << 16);
#endif
}

// ISA 16-bit A-matrix (16x32) per-lane K map: lanes0-15 K={0-7,16-23},
// lanes16-31 K={8-15,24-31}; 2 elems per VGPR, K ascending.
__device__ __forceinline__ int kofA(int e, int hi) {
    int j = e >> 1, p = e & 1;
    return (j < 4) ? (hi * 8 + j * 2 + p) : (16 + hi * 8 + (j - 4) * 2 + p);
}

// ---------------------------------------------------------------------------
// K0: zero the accumulators we atomically add into.
__global__ void init_ws_kernel(float* Z, float* outacc) {
    int t = threadIdx.x;
    if (t < NQ_) Z[t] = 0.0f;
    for (int i = t; i < C_DIM * NQ_; i += 256) outacc[i] = 0.0f;
}

// ---------------------------------------------------------------------------
// K1a: QB[q*128+o] = b1[o] + sum_c W1[o, c] * query[c, q]   (tiny)
__global__ void qb_kernel(const float* __restrict__ W1,
                          const float* __restrict__ query,
                          const float* __restrict__ b1,
                          float* __restrict__ QB) {
    int o = threadIdx.x;  // 128 threads
    for (int q = 0; q < NQ_; ++q) {
        float a = b1[o];
        for (int c = 0; c < C_DIM; ++c)
            a = fmaf(W1[o * 384 + c], query[c * NQ_ + q], a);
        QB[q * C_DIM + o] = a;
    }
}

// ---------------------------------------------------------------------------
// K1b: KBt[k*128+o] = sum_c W1[o, 128+c] * keys[c, k].  LDS-tiled, 16 k/block.
__global__ __launch_bounds__(256) void kb_kernel(const float* __restrict__ W1,
                                                 const float* __restrict__ keys,
                                                 float* __restrict__ KBt) {
    __shared__ float w1s[C_DIM * 33];  // [o][c-panel], pad 33 -> conflict free
    __shared__ float ks[16 * 33];      // [k][c-panel]
    int t = threadIdx.x;
    int kbase = blockIdx.x * 16;
    int o = t & 127, khalf = t >> 7;
    float acc[8] = {0.f, 0.f, 0.f, 0.f, 0.f, 0.f, 0.f, 0.f};

    for (int cp = 0; cp < 4; ++cp) {
        __syncthreads();
        // stage W1k panel (128 o x 32 c), coalesced over c
        for (int j = 0; j < 16; ++j) {
            int idx = t + j * 256;
            int oo = idx >> 5, ci = idx & 31;
            w1s[oo * 33 + ci] = W1[oo * 384 + C_DIM + cp * 32 + ci];
        }
        // stage keys panel (16 k x 32 c), coalesced over k
        for (int j = 0; j < 2; ++j) {
            int idx = t + j * 256;
            int kk = idx & 15, ci = idx >> 4;
            ks[kk * 33 + ci] = keys[(size_t)(cp * 32 + ci) * NK_ + kbase + kk];
        }
        __syncthreads();
        for (int ci = 0; ci < 32; ++ci) {
            float w = w1s[o * 33 + ci];
#pragma unroll
            for (int kk = 0; kk < 8; ++kk)
                acc[kk] = fmaf(w, ks[(khalf * 8 + kk) * 33 + ci], acc[kk]);
        }
    }
#pragma unroll
    for (int kk = 0; kk < 8; ++kk)
        KBt[(size_t)(kbase + khalf * 8 + kk) * C_DIM + o] = acc[kk];
}

// ---------------------------------------------------------------------------
// K1c: pack W1d = W1[:,256:384] into bf16 WMMA A-fragment layout.
// 8 M-tiles x 4 K-tiles; fragment (m,t4): 32 lanes x 16 bf16.
__global__ void pack_w1d_kernel(const float* __restrict__ W1,
                                unsigned* __restrict__ Afrag32) {
    int idx = blockIdx.x * 256 + threadIdx.x;  // 0..1023 = (frag, lane)
    if (idx >= 32 * 32) return;
    int lane = idx & 31, frag = idx >> 5;
    int m = frag >> 2, t4 = frag & 3;
    int hi = lane >> 4, M = m * 16 + (lane & 15);
#pragma unroll
    for (int j = 0; j < 8; ++j) {
        int K0 = t4 * 32 + kofA(2 * j, hi);       // kofA(2j+1) == K0 + 1
        Afrag32[idx * 8 + j] = pack_bf16x2(W1[M * 384 + 2 * C_DIM + K0],
                                           W1[M * 384 + 2 * C_DIM + K0 + 1]);
    }
}

// ---------------------------------------------------------------------------
// K2: the WMMA kernel. 8 waves/block; wave w -> 16 k-columns of one q.
// D(128x16) = W1d(128x128,bf16) @ dist_tile(128x16,bf16), f32 accumulate,
// then + KBt + QB, relu, dot W2, cross-half reduce -> raw scores + block min.
__global__ __launch_bounds__(256) void scores_kernel(
    const float* __restrict__ dist, const float* __restrict__ KBt,
    const float* __restrict__ QB, const unsigned short* __restrict__ Afrag,
    const float* __restrict__ W2, const float* __restrict__ b2,
    float* __restrict__ scores, float* __restrict__ minpart) {
    __shared__ alignas(32) unsigned short sA[32 * 32 * 16];  // 32 KB bf16 frags
    __shared__ float sQB[C_DIM], sW2[C_DIM], sred[256];

    int t = threadIdx.x;
    int q = blockIdx.y;

    // stage all 32 A-fragments; use the CDNA5 ASYNCcnt global->LDS DMA path
#if USE_ASYNC_LDS
    {
        unsigned lds_base =
            (unsigned)(size_t)(__attribute__((address_space(3))) char*)(char*)sA;
        unsigned long long gbase = (unsigned long long)(const char*)Afrag;
#pragma unroll
        for (int i = 0; i < 8; ++i) {
            unsigned off = (unsigned)(t + i * 256) * 16u;
            unsigned loff = lds_base + off;
            unsigned long long gaddr = gbase + off;
            asm volatile("global_load_async_to_lds_b128 %0, %1, off"
                         :
                         : "v"(loff), "v"(gaddr)
                         : "memory");
        }
        asm volatile("s_wait_asynccnt 0" ::: "memory");
    }
#else
    {
        const uint4* s4 = (const uint4*)Afrag;
        uint4* d4 = (uint4*)sA;
#pragma unroll
        for (int i = 0; i < 8; ++i) d4[t + i * 256] = s4[t + i * 256];
    }
#endif
    if (t < C_DIM) { sQB[t] = QB[q * C_DIM + t]; sW2[t] = W2[t]; }
    __syncthreads();

    int wave = t >> 5, lane = t & 31, hi = lane >> 4, n = lane & 15;
    int kcol = blockIdx.x * 128 + wave * 16 + n;
    const float* dcol = dist + (size_t)q * NK_ + kcol;

    // warm the L2/L0 with this lane's KBt epilogue row (512 B, scattered reads
    // later) while the WMMAs run: probe-confirmed -> global_prefetch_b8
    const float* kbrow = KBt + (size_t)kcol * C_DIM;
    __builtin_prefetch(kbrow, 0, 1);
    __builtin_prefetch(kbrow + 64, 0, 1);

    v8f acc[8];
#pragma unroll
    for (int m = 0; m < 8; ++m) acc[m] = (v8f){0.f, 0.f, 0.f, 0.f, 0.f, 0.f, 0.f, 0.f};

    // B fragments: same K map as A (kofA): lanes0-15 K={0-7,16-23}, lanes16-31 K={8-15,24-31}.
    // Coalesced: each element load = 16 contiguous floats per half-wave row.
    v16bf B[4];
#pragma unroll
    for (int t4 = 0; t4 < 4; ++t4) {
        v8ui w;
#pragma unroll
        for (int j = 0; j < 8; ++j) {
            int c0 = t4 * 32 + kofA(2 * j, hi);
            w[j] = pack_bf16x2(dcol[(size_t)c0 * LL_],
                               dcol[(size_t)(c0 + 1) * LL_]);
        }
        B[t4] = __builtin_bit_cast(v16bf, w);
    }

    // 32 WMMAs: full 128(out) x 128(c) x 16(k) per wave
#pragma unroll
    for (int m = 0; m < 8; ++m) {
#pragma unroll
        for (int t4 = 0; t4 < 4; ++t4) {
            v16bf a = *(const v16bf*)(&sA[((m * 4 + t4) * 32 + lane) * 16]);
            acc[m] = __builtin_amdgcn_wmma_f32_16x16x32_bf16(
                false, a, false, B[t4], (short)0, acc[m], false, false);
        }
    }

    // epilogue: + KBt + QB, relu, dot with W2 over this lane's 64 o's
    float partial = 0.f;
#pragma unroll
    for (int m = 0; m < 8; ++m) {
#pragma unroll
        for (int r = 0; r < 8; ++r) {
            int o = m * 16 + hi * 8 + r;   // C-layout: VGPR r, +8 for hi lanes
            float p = acc[m][r] + kbrow[o] + sQB[o];
            p = fmaxf(p, 0.f);
            partial = fmaf(sW2[o], p, partial);
        }
    }
    partial += __shfl_xor(partial, 16);  // combine the two o-halves
    float sc = partial + b2[0];
    if (hi == 0) scores[(size_t)q * NK_ + kcol] = sc;

    // block min partial (for the mask floor)
    sred[t] = sc;
    __syncthreads();
    for (int off = 128; off > 0; off >>= 1) {
        if (t < off) sred[t] = fminf(sred[t], sred[t + off]);
        __syncthreads();
    }
    if (t == 0) minpart[blockIdx.y * gridDim.x + blockIdx.x] = sred[0];
}

// ---------------------------------------------------------------------------
// K2b: floor = min(all raw scores) - 20
__global__ void reduce_min_kernel(const float* __restrict__ minpart, int n,
                                  float* __restrict__ floorv) {
    __shared__ float sred[256];
    int t = threadIdx.x;
    float lm = 3.4e38f;
    for (int i = t; i < n; i += 256) lm = fminf(lm, minpart[i]);
    sred[t] = lm;
    __syncthreads();
    for (int off = 128; off > 0; off >>= 1) {
        if (t < off) sred[t] = fminf(sred[t], sred[t + off]);
        __syncthreads();
    }
    if (t == 0) floorv[0] = sred[0] - 20.0f;
}

// ---------------------------------------------------------------------------
// K3: scores += floor * (!mask) (written back = returned scores), per-q max.
__global__ __launch_bounds__(256) void mask_kernel(
    float* __restrict__ scores, const unsigned char* __restrict__ mask,
    const float* __restrict__ floorv, float* __restrict__ maxpart) {
    __shared__ float sred[256];
    int t = threadIdx.x, q = blockIdx.y, kb = blockIdx.x * 4096;
    float fl = floorv[0];
    float lm = -3.4e38f;
    for (int i = 0; i < 16; ++i) {
        size_t idx = (size_t)q * NK_ + kb + t + i * 256;
        float s = scores[idx];
        if (!mask[idx]) s += fl;
        scores[idx] = s;
        lm = fmaxf(lm, s);
    }
    sred[t] = lm;
    __syncthreads();
    for (int off = 128; off > 0; off >>= 1) {
        if (t < off) sred[t] = fmaxf(sred[t], sred[t + off]);
        __syncthreads();
    }
    if (t == 0) maxpart[q * gridDim.x + blockIdx.x] = sred[0];
}

// K3b: per-q max finalize
__global__ void qmax_kernel(const float* __restrict__ maxpart, int nb,
                            float* __restrict__ qmax) {
    int t = threadIdx.x;
    if (t < NQ_) {
        float m = -3.4e38f;
        for (int i = 0; i < nb; ++i) m = fmaxf(m, maxpart[t * nb + i]);
        qmax[t] = m;
    }
}

// ---------------------------------------------------------------------------
// K4: e = exp(s - qmax), partial Z, and partial out += value @ e (4096-k tile,
// e staged in LDS, coalesced float4 value reads, wave-reduce + f32 atomics).
__global__ __launch_bounds__(256) void softmax_av_kernel(
    const float* __restrict__ scores, const float* __restrict__ value,
    const float* __restrict__ qmax, float* __restrict__ Z,
    float* __restrict__ outacc) {
    __shared__ alignas(16) float e_lds[4096];
    __shared__ float sred[256];
    int t = threadIdx.x, q = blockIdx.y, kb = blockIdx.x * 4096;
    float qm = qmax[q];
    float z = 0.f;
    for (int i = 0; i < 16; ++i) {
        int k = t + i * 256;
        float s = scores[(size_t)q * NK_ + kb + k];
        float e = __expf(s - qm);
        e_lds[k] = e;
        z += e;
    }
    sred[t] = z;
    __syncthreads();
    for (int off = 128; off > 0; off >>= 1) {
        if (t < off) sred[t] += sred[t + off];
        __syncthreads();
    }
    if (t == 0) atomicAdd(&Z[q], sred[0]);

    int wave = t >> 5, lane = t & 31;
    for (int dd = 0; dd < 16; ++dd) {
        int d = wave * 16 + dd;
        const float4* vrow = (const float4*)(value + (size_t)d * NK_ + kb);
        const float4* erow = (const float4*)e_lds;
        float a = 0.f;
        for (int k4 = lane; k4 < 1024; k4 += 32) {
            float4 v = vrow[k4];
            float4 e = erow[k4];
            a = fmaf(v.x, e.x, a);
            a = fmaf(v.y, e.y, a);
            a = fmaf(v.z, e.z, a);
            a = fmaf(v.w, e.w, a);
        }
#pragma unroll
        for (int off = 16; off > 0; off >>= 1) a += __shfl_xor(a, off);
        if (lane == 0) atomicAdd(&outacc[q * C_DIM + d], a);
    }
}

// K5: out[d, q] = outacc[q, d] / Z[q]   (out shape (1, C, NQ) -> d*NQ + q)
__global__ void finalize_out_kernel(const float* __restrict__ outacc,
                                    const float* __restrict__ Z,
                                    float* __restrict__ out) {
    int d = threadIdx.x;  // 128
    for (int q = 0; q < NQ_; ++q) out[d * NQ_ + q] = outacc[q * C_DIM + d] / Z[q];
}

// ---------------------------------------------------------------------------
extern "C" void kernel_launch(void* const* d_in, const int* in_sizes, int n_in,
                              void* d_out, int out_size, void* d_ws,
                              size_t ws_size, hipStream_t stream) {
    (void)in_sizes; (void)n_in; (void)out_size; (void)ws_size;
    const float* query = (const float*)d_in[0];
    const float* keys  = (const float*)d_in[1];
    const float* value = (const float*)d_in[2];
    const float* dist  = (const float*)d_in[3];
    const unsigned char* mask = (const unsigned char*)d_in[4];  // bool tensor
    const float* W1 = (const float*)d_in[5];
    const float* b1 = (const float*)d_in[6];
    const float* W2 = (const float*)d_in[7];
    const float* b2 = (const float*)d_in[8];

    float* out    = (float*)d_out;          // [0 .. 1023] : (1, C, NQ)
    float* scores = out + C_DIM * NQ_;      // [1024 ..]   : (1, NQ, NK)

    // workspace carve-up (256B aligned); KBt dominates at ~16.8 MB
    char* ws = (char*)d_ws;
    size_t off = 0;
    auto carve = [&](size_t bytes) {
        size_t cur = off;
        off += (bytes + 255) & ~(size_t)255;
        return cur;
    };
    float* QB      = (float*)(ws + carve((size_t)NQ_ * C_DIM * 4));
    float* KBt     = (float*)(ws + carve((size_t)NK_ * C_DIM * 4));
    unsigned short* Afrag = (unsigned short*)(ws + carve((size_t)32 * 32 * 16 * 2));
    float* minpart = (float*)(ws + carve((size_t)2048 * 4));
    float* maxpart = (float*)(ws + carve((size_t)64 * 4));
    float* floorv  = (float*)(ws + carve(4));
    float* qmax    = (float*)(ws + carve((size_t)NQ_ * 4));
    float* Z       = (float*)(ws + carve((size_t)NQ_ * 4));
    float* outacc  = (float*)(ws + carve((size_t)NQ_ * C_DIM * 4));

    init_ws_kernel<<<1, 256, 0, stream>>>(Z, outacc);
    qb_kernel<<<1, 128, 0, stream>>>(W1, query, b1, QB);
    kb_kernel<<<NK_ / 16, 256, 0, stream>>>(W1, keys, KBt);
    pack_w1d_kernel<<<4, 256, 0, stream>>>(W1, (unsigned*)Afrag);

    scores_kernel<<<dim3(NK_ / 128, NQ_), 256, 0, stream>>>(
        dist, KBt, QB, Afrag, W2, b2, scores, minpart);

    reduce_min_kernel<<<1, 256, 0, stream>>>(minpart, (NK_ / 128) * NQ_, floorv);
    mask_kernel<<<dim3(NK_ / 4096, NQ_), 256, 0, stream>>>(scores, mask, floorv,
                                                           maxpart);
    qmax_kernel<<<1, 64, 0, stream>>>(maxpart, NK_ / 4096, qmax);
    softmax_av_kernel<<<dim3(NK_ / 4096, NQ_), 256, 0, stream>>>(scores, value,
                                                                 qmax, Z, outacc);
    finalize_out_kernel<<<1, 128, 0, stream>>>(outacc, Z, out);
}
